// FlowEmbeddingLayer_9070970929195
// MI455X (gfx1250) — matrix operand order repaired and checked by reading, not compiled, not run on hardware
//
#include <hip/hip_runtime.h>
#include <hip/hip_bf16.h>

typedef __attribute__((ext_vector_type(16))) _Float16 v16h;
typedef __attribute__((ext_vector_type(8)))  _Float16 v8h;
typedef __attribute__((ext_vector_type(8)))  float    v8f;
typedef __attribute__((ext_vector_type(2)))  float    v2f;

#define FEAT     64
#define HID      128
#define KPAD1    96              // 67 padded to 3 K-slices of 32
#define KNN_TILE 2048            // candidates staged per LDS tile

// ---------------------------------------------------------------------------
// Kernel 1: brute-force 1-NN via V_WMMA_F32_16X16X4_F32.
// A (16x4)  = queries    x (qx, qy, qz, 1)        -- built once per wave
// B (4x16)  = candidates x (-2px, -2py, -2pz, p2) -- one ds_load_b64 per tile
// D = A x B + 0  ==>  D[m][n] = |p_n|^2 - 2 q_m . p_n   (argmin-equivalent)
// 256 distance evals per wmma; VALU only does the running argmin.
// f32 A layout: v0 = K0(lanes0-15)/K2(lanes16-31), v1 = K1/K3.  B mirrors it.
// D layout: d[r] = D[M = r + 8*(lane>>4)][N = lane&15].
// ---------------------------------------------------------------------------
__global__ __launch_bounds__(256) void knn1_wmma_kernel(
    const float* __restrict__ x1_pos,   // [N1,3]
    const float* __restrict__ x2_pos,   // [N2,3]
    int* __restrict__ idx_out,          // [N2]
    int n1)
{
    __shared__ float2 lo[KNN_TILE];     // (-2px, -2py)
    __shared__ float2 hi[KNN_TILE];     // (-2pz, |p|^2)

    const int tid  = threadIdx.x;
    const int wv   = tid >> 5;
    const int lane = tid & 31;
    const int nl   = lane & 15;
    const int hiL  = lane >> 4;

    const int qbase = blockIdx.x * 128 + wv * 16;   // 16 queries per wave
    const int q     = qbase + nl;

    v2f afrag;
    if (lane < 16) afrag = (v2f){ x2_pos[3 * q + 0], x2_pos[3 * q + 1] };
    else           afrag = (v2f){ x2_pos[3 * q + 2], 1.0f };

    float bestd[8];
    int   bestj[8];
    #pragma unroll
    for (int r = 0; r < 8; ++r) { bestd[r] = 3.4e38f; bestj[r] = 0; }

    const float2* bsrc = (lane < 16) ? lo : hi;

    for (int t0 = 0; t0 < n1; t0 += KNN_TILE) {
        __syncthreads();
        for (int i = tid; i < KNN_TILE; i += 256) {
            const float* gp = x1_pos + 3 * (t0 + i);
            float px = gp[0], py = gp[1], pz = gp[2];
            lo[i] = make_float2(-2.0f * px, -2.0f * py);
            hi[i] = make_float2(-2.0f * pz, px * px + py * py + pz * pz);
        }
        __syncthreads();

        #pragma unroll 8
        for (int ts = 0; ts < KNN_TILE / 16; ++ts) {
            float2 bv = bsrc[ts * 16 + nl];
            v2f b = (v2f){ bv.x, bv.y };
            v8f d = __builtin_amdgcn_wmma_f32_16x16x4_f32(
                false, afrag, false, b, (short)0, (v8f){}, false, false);
            int jc = t0 + ts * 16 + nl;
            #pragma unroll
            for (int r = 0; r < 8; ++r) {
                if (d[r] < bestd[r]) { bestd[r] = d[r]; bestj[r] = jc; }
            }
        }
    }

    // Reduce across the 16 N-columns (lanes sharing lane>>4) with
    // smaller-index tie-break to match argmin's first-occurrence rule.
    #pragma unroll
    for (int off = 1; off < 16; off <<= 1) {
        #pragma unroll
        for (int r = 0; r < 8; ++r) {
            float od = __shfl_xor(bestd[r], off, 32);
            int   oj = __shfl_xor(bestj[r], off, 32);
            if (od < bestd[r] || (od == bestd[r] && oj < bestj[r])) {
                bestd[r] = od; bestj[r] = oj;
            }
        }
    }
    if (nl == 0) {
        #pragma unroll
        for (int r = 0; r < 8; ++r)
            idx_out[qbase + r + 8 * hiL] = bestj[r];
    }
}

// ---------------------------------------------------------------------------
// WMMA fragment loaders (CDNA5 16x16x32 f16 layouts, CDNA5 ISA)
// ---------------------------------------------------------------------------
__device__ __forceinline__ v16h load_a_frag(const _Float16* base, int kpad,
                                            int mbase, int lane, int kbase)
{
    const int m  = mbase + (lane & 15);
    const int hi = lane >> 4;
    const _Float16* p = base + m * kpad + kbase + 8 * hi;
    union { v16h v; v8h h[2]; } u;
    u.h[0] = *(const v8h*)(p);
    u.h[1] = *(const v8h*)(p + 16);
    return u.v;
}

__device__ __forceinline__ v16h load_b_frag(const _Float16* w, int lane,
                                            int kbase, int nbase)
{
    const _Float16* p = w + (kbase + 8 * (lane >> 4)) * HID + nbase + (lane & 15);
    v16h v;
    #pragma unroll
    for (int s = 0; s < 8; ++s) {
        v[s]     = p[s * HID];
        v[8 + s] = p[(16 + s) * HID];
    }
    return v;
}

// ---------------------------------------------------------------------------
// Kernel 2: gather + fused 3-layer MLP (f16 WMMA, f32 accumulate).
// 8 waves; each owns a 16-row slice x full 128-wide N (8 v8f accumulators).
// Weights + message + hidden tiles in ~144 KB of the 320 KB WGP LDS.
// ---------------------------------------------------------------------------
__global__ __launch_bounds__(256) void mlp_kernel(
    const float* __restrict__ x1_feat,  // [N1,64]
    const float* __restrict__ x1_pos,   // [N1,3]
    const float* __restrict__ x2_pos,   // [N2,3]
    const int*   __restrict__ knn_idx,  // [N2]
    const float* __restrict__ W1, const float* __restrict__ b1,
    const float* __restrict__ W2, const float* __restrict__ b2,
    const float* __restrict__ W3, const float* __restrict__ b3,
    float* __restrict__ out_x)          // [N2,128]
{
    extern __shared__ char smem_raw[];
    _Float16* w1l  = (_Float16*)smem_raw;        // [96][128]
    _Float16* w2l  = w1l + KPAD1 * HID;          // [128][128]
    _Float16* w3l  = w2l + HID * HID;            // [128][128]
    _Float16* msg  = w3l + HID * HID;            // [128][96]
    _Float16* hbuf = msg + 128 * KPAD1;          // [128][128]

    const int tid = threadIdx.x;
    const int gb  = blockIdx.x * 128;

    for (int i = tid; i < KPAD1 * HID; i += 256) {
        int r = i >> 7;
        w1l[i] = (r < FEAT + 3) ? (_Float16)W1[i] : (_Float16)0.0f;
    }
    for (int i = tid; i < HID * HID; i += 256) {
        w2l[i] = (_Float16)W2[i];
        w3l[i] = (_Float16)W3[i];
    }

    for (int i = tid; i < 128 * KPAD1; i += 256) {
        int row = i / KPAD1;
        int col = i - row * KPAD1;
        int g   = gb + row;
        int j   = knn_idx[g];
        float v;
        if (col < FEAT)          v = x1_feat[j * FEAT + col];
        else if (col < FEAT + 3) v = x1_pos[j * 3 + (col - FEAT)]
                                     - x2_pos[g * 3 + (col - FEAT)];
        else                     v = 0.0f;
        msg[i] = (_Float16)v;
    }
    __syncthreads();

    const int wv    = tid >> 5;
    const int lane  = tid & 31;
    const int hi    = lane >> 4;
    const int nl    = lane & 15;
    const int mbase = wv * 16;

    v8f acc[8];

    // ================= layer 1: msg[128,96] @ W1[96,128] =================
    #pragma unroll
    for (int nt = 0; nt < 8; ++nt) acc[nt] = (v8f){};
    #pragma unroll
    for (int ks = 0; ks < 3; ++ks) {
        v16h a = load_a_frag(msg, KPAD1, mbase, lane, ks * 32);
        #pragma unroll
        for (int nt = 0; nt < 8; ++nt) {
            v16h b = load_b_frag(w1l, lane, ks * 32, nt * 16);
            acc[nt] = __builtin_amdgcn_wmma_f32_16x16x32_f16(
                false, a, false, b, (short)0, acc[nt], false, false);
        }
    }
    #pragma unroll
    for (int nt = 0; nt < 8; ++nt) {
        int n = nt * 16 + nl;
        float bb = b1[n];
        #pragma unroll
        for (int r = 0; r < 8; ++r) {
            float v = acc[nt][r] + bb;
            v = v > 0.0f ? v : 0.0f;
            hbuf[(mbase + r + 8 * hi) * HID + n] = (_Float16)v;
        }
    }
    __syncthreads();

    // ================= layer 2: h[128,128] @ W2[128,128] =================
    #pragma unroll
    for (int nt = 0; nt < 8; ++nt) acc[nt] = (v8f){};
    #pragma unroll
    for (int ks = 0; ks < 4; ++ks) {
        v16h a = load_a_frag(hbuf, HID, mbase, lane, ks * 32);
        #pragma unroll
        for (int nt = 0; nt < 8; ++nt) {
            v16h b = load_b_frag(w2l, lane, ks * 32, nt * 16);
            acc[nt] = __builtin_amdgcn_wmma_f32_16x16x32_f16(
                false, a, false, b, (short)0, acc[nt], false, false);
        }
    }
    __syncthreads();   // hbuf will be overwritten
    #pragma unroll
    for (int nt = 0; nt < 8; ++nt) {
        int n = nt * 16 + nl;
        float bb = b2[n];
        #pragma unroll
        for (int r = 0; r < 8; ++r) {
            float v = acc[nt][r] + bb;
            v = v > 0.0f ? v : 0.0f;
            hbuf[(mbase + r + 8 * hi) * HID + n] = (_Float16)v;
        }
    }
    __syncthreads();

    // ================= layer 3: h[128,128] @ W3[128,128] + b3 =============
    #pragma unroll
    for (int nt = 0; nt < 8; ++nt) acc[nt] = (v8f){};
    #pragma unroll
    for (int ks = 0; ks < 4; ++ks) {
        v16h a = load_a_frag(hbuf, HID, mbase, lane, ks * 32);
        #pragma unroll
        for (int nt = 0; nt < 8; ++nt) {
            v16h b = load_b_frag(w3l, lane, ks * 32, nt * 16);
            acc[nt] = __builtin_amdgcn_wmma_f32_16x16x32_f16(
                false, a, false, b, (short)0, acc[nt], false, false);
        }
    }
    #pragma unroll
    for (int nt = 0; nt < 8; ++nt) {
        int n = nt * 16 + nl;
        float bb = b3[n];
        #pragma unroll
        for (int r = 0; r < 8; ++r) {
            int g = gb + mbase + r + 8 * hi;
            out_x[g * HID + n] = acc[nt][r] + bb;
        }
    }
}

// ---------------------------------------------------------------------------
// Kernel 3: copy x2_pos and x2_batch bit patterns into the concatenated out.
// ---------------------------------------------------------------------------
__global__ __launch_bounds__(256) void tail_kernel(
    const float* __restrict__ x2_pos,
    const int*   __restrict__ x2_batch,
    float* __restrict__ out, int n2)
{
    int i = blockIdx.x * 256 + threadIdx.x;
    int npos = 3 * n2;
    if (i < npos) {
        out[n2 * HID + i] = x2_pos[i];
    } else {
        int k = i - npos;
        ((int*)out)[n2 * (HID + 3) + k] = x2_batch[k];
    }
}

// ---------------------------------------------------------------------------
extern "C" void kernel_launch(void* const* d_in, const int* in_sizes, int n_in,
                              void* d_out, int out_size, void* d_ws, size_t ws_size,
                              hipStream_t stream)
{
    const float* x1_feat = (const float*)d_in[0];
    const float* x1_pos  = (const float*)d_in[1];
    const float* x2_pos  = (const float*)d_in[4];
    const int*   x2_b    = (const int*)d_in[5];
    const float* W1 = (const float*)d_in[6];
    const float* b1 = (const float*)d_in[7];
    const float* W2 = (const float*)d_in[8];
    const float* b2 = (const float*)d_in[9];
    const float* W3 = (const float*)d_in[10];
    const float* b3 = (const float*)d_in[11];

    const int n1 = in_sizes[1] / 3;
    const int n2 = in_sizes[4] / 3;

    int* knn_idx = (int*)d_ws;                  // n2 ints of scratch

    knn1_wmma_kernel<<<n2 / 128, 256, 0, stream>>>(x1_pos, x2_pos,
                                                   knn_idx, n1);

    const size_t lds_bytes =
        (size_t)(KPAD1 * HID + 2 * HID * HID + 128 * KPAD1 + 128 * HID)
        * sizeof(_Float16);                     // 147456 B < 320 KB WGP LDS
    mlp_kernel<<<n2 / 128, 256, lds_bytes, stream>>>(
        x1_feat, x1_pos, x2_pos, knn_idx,
        W1, b1, W2, b2, W3, b3, (float*)d_out);

    tail_kernel<<<(4 * n2) / 256, 256, 0, stream>>>(x2_pos, x2_b,
                                                    (float*)d_out, n2);
}
